// Attention_23270132810385
// MI455X (gfx1250) — matrix operand order repaired and checked by reading, not compiled, not run on hardware
//
#include <hip/hip_runtime.h>
#include <hip/hip_bf16.h>

typedef __attribute__((ext_vector_type(16))) _Float16 v16h;
typedef __attribute__((ext_vector_type(8)))  _Float16 v8h;
typedef __attribute__((ext_vector_type(8)))  float    v8f;

#define D_MODEL 512
#define N_HEADS 8
#define HEAD_DIM 64
#define BATCH 4
#define SEQ 2048
#define BS (BATCH * SEQ) /* 8192 */

// CDNA5 async global->LDS bulk copy (ASYNCcnt path). One-line revert switch.
#define USE_ASYNC_LDS 1

typedef __attribute__((address_space(3))) _Float16 lds_f16;

static __device__ __forceinline__ void copy_b128_to_lds(const _Float16* gsrc,
                                                        _Float16* ldst) {
#if USE_ASYNC_LDS
  unsigned laddr = (unsigned)(unsigned long long)(lds_f16*)ldst;
  unsigned long long gaddr = (unsigned long long)gsrc;
  asm volatile("global_load_async_to_lds_b128 %0, %1, off"
               :: "v"(laddr), "v"(gaddr) : "memory");
#else
  *(v8h*)ldst = *(const v8h*)gsrc;
#endif
}

static __device__ __forceinline__ void async_copy_wait() {
#if USE_ASYNC_LDS
  asm volatile("s_wait_asynccnt 0x0" ::: "memory");
#endif
}

static __device__ __forceinline__ v8f wmma_f16(v16h a, v16h b, v8f c) {
  return __builtin_amdgcn_wmma_f32_16x16x32_f16(
      false, a, false, b, (short)0, c, false, false);
}

static __device__ __forceinline__ v16h cat8(v8h lo, v8h hi) {
  return __builtin_shufflevector(lo, hi, 0, 1, 2, 3, 4, 5, 6, 7,
                                 8, 9, 10, 11, 12, 13, 14, 15);
}

// A fragment (16x32 f16, this file's contiguous k order, not the ISA table order): lane = hi*16+row;
// elems 0..15 -> K = hi*16+e  => two aligned 16B runs
static __device__ __forceinline__ v16h load_a_frag(const _Float16* row0, int lane) {
  const int hi = lane >> 4;
  return cat8(*(const v8h*)(row0 + hi * 16), *(const v8h*)(row0 + hi * 16 + 8));
}
// B fragment from an N-major (K-contiguous) tile: 16 contiguous K values per lane.
static __device__ __forceinline__ v16h load_b_frag_nmajor(const _Float16* col0, int lane) {
  const int hi = lane >> 4;
  return cat8(*(const v8h*)(col0 + hi * 16), *(const v8h*)(col0 + hi * 16 + 8));
}

// ---------------- kernel 1a: fp32 -> f16 conversion of x ----------------
__global__ __launch_bounds__(256) void cvt_x(const float* __restrict__ x,
                                             _Float16* __restrict__ xh) {
  const int stride = gridDim.x * blockDim.x;
  for (int i = blockIdx.x * blockDim.x + threadIdx.x; i < BS * D_MODEL; i += stride)
    xh[i] = (_Float16)x[i];
}

// ---------------- kernel 1b: fused transpose + fp32->f16 of weights --------
// Produces wT[n][k] (N-major) so GEMM B-tiles are K-contiguous per column.
struct WPtrs {
  const float* w[4];
  _Float16* wT[4];
};

__global__ __launch_bounds__(256) void cvt_wT(WPtrs p) {
  __shared__ _Float16 tile[32][33];
  const float* __restrict__ w = p.w[blockIdx.z];
  _Float16* __restrict__ wT = p.wT[blockIdx.z];
  const int rb = blockIdx.x * 32, cb = blockIdx.y * 32;
  const int tr = threadIdx.x >> 5, tc = threadIdx.x & 31;
#pragma unroll
  for (int i = 0; i < 4; ++i)
    tile[tr + i * 8][tc] = (_Float16)w[(size_t)(rb + tr + i * 8) * D_MODEL + cb + tc];
  __syncthreads();
#pragma unroll
  for (int i = 0; i < 4; ++i)
    wT[(size_t)(cb + tr + i * 8) * D_MODEL + rb + tc] = tile[tc][tr + i * 8];
}

// ---------------- kernel 2: fused QKV projection GEMM ----------------
// Block tile 128(M) x 64(N); 8 waves, each computing a 32x32 tile with 2x2
// WMMA register blocking (2 ds_load_b128 per WMMA).
// Q,K out: head-major [B,H,S,Hd] f16. V out: d-major [B,H,Hd,S] f16 so the
// attention kernel can async-stage transposed V tiles contiguously.
struct QKVPtrs {
  const _Float16* WT[3];
  const float* bias[3];
  _Float16* out[3];
};

__global__ __launch_bounds__(256) void gemm_qkv(const _Float16* __restrict__ xh,
                                                QKVPtrs p) {
  __shared__ _Float16 As[2][128][40];  // 128 rows x 32 K
  __shared__ _Float16 BsT[2][64][40];  // 64 cols x 32 K (N-major)

  const int z = blockIdx.z;
  const _Float16* __restrict__ WT = p.WT[z];
  const float* __restrict__ bias = p.bias[z];
  _Float16* __restrict__ out = p.out[z];

  const int tid = threadIdx.x;
  const int lane = tid & 31, wave = tid >> 5;
  const int hi = lane >> 4, ln = lane & 15;
  const int msub = wave & 3, npair = wave >> 2;
  const int rowbase = blockIdx.x * 128;
  const int colbase = blockIdx.y * 64;

  // staging: A 128x32 = 512 chunks (2/thread); B 64x32 = 256 chunks (1/thread)
  const int bc = tid >> 2, bk8 = (tid & 3) * 8;

  auto stage = [&](int buf, int k0) {
#pragma unroll
    for (int j = 0; j < 2; ++j) {
      int ch = tid + j * 256;
      int rr = ch >> 2, cc8 = (ch & 3) * 8;
      copy_b128_to_lds(&xh[(size_t)(rowbase + rr) * D_MODEL + k0 + cc8],
                       &As[buf][rr][cc8]);
    }
    copy_b128_to_lds(&WT[(size_t)(colbase + bc) * D_MODEL + k0 + bk8],
                     &BsT[buf][bc][bk8]);
  };

  v8f acc[2][2];
  const v8f vzero = {};
#pragma unroll
  for (int i = 0; i < 2; ++i)
#pragma unroll
    for (int j = 0; j < 2; ++j) acc[i][j] = vzero;

  stage(0, 0);
  for (int step = 0; step < D_MODEL / 32; ++step) {
    const int buf = step & 1;
    async_copy_wait();
    __syncthreads();
    if (step + 1 < D_MODEL / 32) stage(buf ^ 1, (step + 1) * 32); // overlap

    v16h a0 = load_a_frag(&As[buf][msub * 32 + ln][0], lane);
    v16h a1 = load_a_frag(&As[buf][msub * 32 + 16 + ln][0], lane);
    v16h b0 = load_b_frag_nmajor(&BsT[buf][npair * 32 + ln][0], lane);
    v16h b1 = load_b_frag_nmajor(&BsT[buf][npair * 32 + 16 + ln][0], lane);
    acc[0][0] = wmma_f16(a0, b0, acc[0][0]);
    acc[0][1] = wmma_f16(a0, b1, acc[0][1]);
    acc[1][0] = wmma_f16(a1, b0, acc[1][0]);
    acc[1][1] = wmma_f16(a1, b1, acc[1][1]);
  }

  // epilogue: +bias, scatter
#pragma unroll
  for (int tm = 0; tm < 2; ++tm) {
#pragma unroll
    for (int tn = 0; tn < 2; ++tn) {
      v8f a = acc[tm][tn];
      int col = colbase + npair * 32 + tn * 16 + ln;
      int h = col >> 6, hd = col & 63;
      float bval = bias[col];
#pragma unroll
      for (int r = 0; r < 8; ++r) {
        int m = rowbase + msub * 32 + tm * 16 + r + hi * 8; // C: M = r + 8*hi
        int bb = m >> 11, s = m & (SEQ - 1);
        _Float16 val = (_Float16)(a[r] + bval);
        if (z == 2) // V: d-major [B,H,Hd,S]
          out[((size_t)(bb * N_HEADS + h) * HEAD_DIM + hd) * SEQ + s] = val;
        else        // Q,K: [B,H,S,Hd]
          out[(((size_t)(bb * N_HEADS + h) * SEQ + s) << 6) + hd] = val;
      }
    }
  }
}

// ---------------- kernel 3: flash attention ----------------
__global__ __launch_bounds__(256) void flash_attn(const _Float16* __restrict__ Qh,
                                                  const _Float16* __restrict__ Kh,
                                                  const _Float16* __restrict__ Vt,
                                                  _Float16* __restrict__ Oh) {
  __shared__ _Float16 Ks[2][32][72];   // 32 keys x 64 d (K-major)
  __shared__ _Float16 VsT[2][64][40];  // 64 d x 32 keys (d-major)
  __shared__ _Float16 Ps[8][16][40];   // per-wave P staging 16x32

  const int tid = threadIdx.x, lane = tid & 31, wave = tid >> 5;
  const int hi = lane >> 4, ln = lane & 15;
  const int bh = blockIdx.y; // b*H + h
  const int b = bh >> 3, h = bh & 7;
  const int qrow0 = blockIdx.x * 128 + wave * 16;

  const _Float16* Qp = Qh + (size_t)bh * SEQ * HEAD_DIM;
  const _Float16* Kp = Kh + (size_t)bh * SEQ * HEAD_DIM;
  const _Float16* Vtp = Vt + (size_t)bh * HEAD_DIM * SEQ;

  // staging coordinates
  const int kr = tid >> 3, kc8 = (tid & 7) * 8; // K tile 32 keys x 64 d
  const int vd = tid >> 2, vs8 = (tid & 3) * 8; // V tile 64 d x 32 keys

  auto stage = [&](int buf, int kt) {
    copy_b128_to_lds(&Kp[(size_t)(kt * 32 + kr) * HEAD_DIM + kc8],
                     &Ks[buf][kr][kc8]);
    copy_b128_to_lds(&Vtp[(size_t)vd * SEQ + kt * 32 + vs8],
                     &VsT[buf][vd][vs8]);
  };

  // Q fragments (16 rows x 64 d), pre-scaled by 1/sqrt(Hd)=0.125
  v16h aq0, aq1;
  {
    const _Float16* qr = Qp + (size_t)(qrow0 + ln) * HEAD_DIM;
    v8h q0 = *(const v8h*)(qr + hi * 16);
    v8h q1 = *(const v8h*)(qr + hi * 16 + 8);
    v8h q2 = *(const v8h*)(qr + 32 + hi * 16);
    v8h q3 = *(const v8h*)(qr + 40 + hi * 16);
    const _Float16 qs = (_Float16)0.125f;
    q0 = q0 * qs; q1 = q1 * qs; q2 = q2 * qs; q3 = q3 * qs;
    aq0 = cat8(q0, q1);
    aq1 = cat8(q2, q3);
  }

  const v8f vzero = {};
  float m_run[8], l_run[8];
#pragma unroll
  for (int r = 0; r < 8; ++r) { m_run[r] = -3.0e38f; l_run[r] = 0.0f; }
  v8f accO[4];
#pragma unroll
  for (int n = 0; n < 4; ++n) accO[n] = vzero;

  stage(0, 0);
  for (int kt = 0; kt < SEQ / 32; ++kt) {
    const int buf = kt & 1;
    async_copy_wait();
    __syncthreads();
    if (kt + 1 < SEQ / 32) stage(buf ^ 1, kt + 1); // overlap with compute

    // S = Q * K^T : two 16-key n-tiles, K-dim = Hd = 64 -> 2 WMMA steps each
    v8f sc0 = vzero, sc1 = vzero;
#pragma unroll
    for (int ks = 0; ks < 2; ++ks) {
      const _Float16* kr0 = &Ks[buf][ln][ks * 32 + hi * 16];
      const _Float16* kr1 = &Ks[buf][16 + ln][ks * 32 + hi * 16];
      v16h bk0 = cat8(*(const v8h*)kr0, *(const v8h*)(kr0 + 8));
      v16h bk1 = cat8(*(const v8h*)kr1, *(const v8h*)(kr1 + 8));
      sc0 = wmma_f16(ks ? aq1 : aq0, bk0, sc0);
      sc1 = wmma_f16(ks ? aq1 : aq0, bk1, sc1);
    }

    // online softmax; rows live across 16-lane half-groups (shfl masks 1..8)
    float alpha[8], pa[8], pb[8];
#pragma unroll
    for (int r = 0; r < 8; ++r) {
      float mx = fmaxf(sc0[r], sc1[r]);
#pragma unroll
      for (int msk = 1; msk <= 8; msk <<= 1)
        mx = fmaxf(mx, __shfl_xor(mx, msk, 32));
      float mn = fmaxf(m_run[r], mx);
      alpha[r] = __expf(m_run[r] - mn);
      m_run[r] = mn;
      pa[r] = __expf(sc0[r] - mn);
      pb[r] = __expf(sc1[r] - mn);
      float rs = pa[r] + pb[r];
#pragma unroll
      for (int msk = 1; msk <= 8; msk <<= 1)
        rs += __shfl_xor(rs, msk, 32);
      l_run[r] = l_run[r] * alpha[r] + rs;
    }
#pragma unroll
    for (int n = 0; n < 4; ++n)
#pragma unroll
      for (int r = 0; r < 8; ++r) accO[n][r] = accO[n][r] * alpha[r];

    // restage P (C layout) -> LDS -> A layout (wave-private, b128 reads)
    _Float16* myP = &Ps[wave][0][0];
#pragma unroll
    for (int r = 0; r < 8; ++r) {
      int m = r + hi * 8;
      myP[m * 40 + ln] = (_Float16)pa[r];
      myP[m * 40 + 16 + ln] = (_Float16)pb[r];
    }
    v16h ap = load_a_frag(&Ps[wave][ln][0], lane);

    // O += P * V : 4 n-tiles over Hd, V read d-major (contiguous)
#pragma unroll
    for (int n = 0; n < 4; ++n) {
      v16h bv = load_b_frag_nmajor(&VsT[buf][n * 16 + ln][0], lane);
      accO[n] = wmma_f16(ap, bv, accO[n]);
    }
  }

  // finalize: O /= l ; store f16 row-major [B*S, 512]
#pragma unroll
  for (int n = 0; n < 4; ++n) {
#pragma unroll
    for (int r = 0; r < 8; ++r) {
      int m = r + hi * 8;
      int qr = qrow0 + m;
      float inv = 1.0f / l_run[r];
      Oh[(size_t)(b * SEQ + qr) * D_MODEL + h * HEAD_DIM + n * 16 + ln] =
          (_Float16)(accO[n][r] * inv);
    }
  }
}

// ---------------- kernel 4: output projection ----------------
__global__ __launch_bounds__(256) void gemm_out(const _Float16* __restrict__ oh,
                                                const _Float16* __restrict__ WT,
                                                const float* __restrict__ bias,
                                                float* __restrict__ out) {
  __shared__ _Float16 As[2][128][40];
  __shared__ _Float16 BsT[2][64][40];

  const int tid = threadIdx.x;
  const int lane = tid & 31, wave = tid >> 5;
  const int hi = lane >> 4, ln = lane & 15;
  const int msub = wave & 3, npair = wave >> 2;
  const int rowbase = blockIdx.x * 128;
  const int colbase = blockIdx.y * 64;

  const int bc = tid >> 2, bk8 = (tid & 3) * 8;

  auto stage = [&](int buf, int k0) {
#pragma unroll
    for (int j = 0; j < 2; ++j) {
      int ch = tid + j * 256;
      int rr = ch >> 2, cc8 = (ch & 3) * 8;
      copy_b128_to_lds(&oh[(size_t)(rowbase + rr) * D_MODEL + k0 + cc8],
                       &As[buf][rr][cc8]);
    }
    copy_b128_to_lds(&WT[(size_t)(colbase + bc) * D_MODEL + k0 + bk8],
                     &BsT[buf][bc][bk8]);
  };

  v8f acc[2][2];
  const v8f vzero = {};
#pragma unroll
  for (int i = 0; i < 2; ++i)
#pragma unroll
    for (int j = 0; j < 2; ++j) acc[i][j] = vzero;

  stage(0, 0);
  for (int step = 0; step < D_MODEL / 32; ++step) {
    const int buf = step & 1;
    async_copy_wait();
    __syncthreads();
    if (step + 1 < D_MODEL / 32) stage(buf ^ 1, (step + 1) * 32);

    v16h a0 = load_a_frag(&As[buf][msub * 32 + ln][0], lane);
    v16h a1 = load_a_frag(&As[buf][msub * 32 + 16 + ln][0], lane);
    v16h b0 = load_b_frag_nmajor(&BsT[buf][npair * 32 + ln][0], lane);
    v16h b1 = load_b_frag_nmajor(&BsT[buf][npair * 32 + 16 + ln][0], lane);
    acc[0][0] = wmma_f16(a0, b0, acc[0][0]);
    acc[0][1] = wmma_f16(a0, b1, acc[0][1]);
    acc[1][0] = wmma_f16(a1, b0, acc[1][0]);
    acc[1][1] = wmma_f16(a1, b1, acc[1][1]);
  }

#pragma unroll
  for (int tm = 0; tm < 2; ++tm) {
#pragma unroll
    for (int tn = 0; tn < 2; ++tn) {
      v8f a = acc[tm][tn];
      int col = colbase + npair * 32 + tn * 16 + ln;
      float bval = bias[col];
#pragma unroll
      for (int r = 0; r < 8; ++r) {
        int m = rowbase + msub * 32 + tm * 16 + r + hi * 8;
        out[(size_t)m * D_MODEL + col] = a[r] + bval;
      }
    }
  }
}

// ---------------- launcher ----------------
extern "C" void kernel_launch(void* const* d_in, const int* in_sizes, int n_in,
                              void* d_out, int out_size, void* d_ws, size_t ws_size,
                              hipStream_t stream) {
  const float* x  = (const float*)d_in[0];
  const float* wq = (const float*)d_in[1];
  const float* bq = (const float*)d_in[2];
  const float* wk = (const float*)d_in[3];
  const float* bk = (const float*)d_in[4];
  const float* wv = (const float*)d_in[5];
  const float* bv = (const float*)d_in[6];
  const float* wo = (const float*)d_in[7];
  const float* bo = (const float*)d_in[8];
  float* out = (float*)d_out;

  char* ws = (char*)d_ws;
  size_t off = 0;
  auto alloc = [&](size_t bytes) {
    void* p = ws + off;
    off += (bytes + 255) & ~(size_t)255;
    return p;
  };
  _Float16* xh   = (_Float16*)alloc((size_t)BS * D_MODEL * 2);
  _Float16* wqT  = (_Float16*)alloc((size_t)D_MODEL * D_MODEL * 2);
  _Float16* wkT  = (_Float16*)alloc((size_t)D_MODEL * D_MODEL * 2);
  _Float16* wvT  = (_Float16*)alloc((size_t)D_MODEL * D_MODEL * 2);
  _Float16* woT  = (_Float16*)alloc((size_t)D_MODEL * D_MODEL * 2);
  _Float16* qh   = (_Float16*)alloc((size_t)BS * D_MODEL * 2);
  _Float16* kh   = (_Float16*)alloc((size_t)BS * D_MODEL * 2);
  _Float16* vt   = (_Float16*)alloc((size_t)BS * D_MODEL * 2);
  _Float16* ohb  = (_Float16*)alloc((size_t)BS * D_MODEL * 2);

  cvt_x<<<2048, 256, 0, stream>>>(x, xh);

  WPtrs wp;
  wp.w[0] = wq; wp.w[1] = wk; wp.w[2] = wv; wp.w[3] = wo;
  wp.wT[0] = wqT; wp.wT[1] = wkT; wp.wT[2] = wvT; wp.wT[3] = woT;
  cvt_wT<<<dim3(D_MODEL / 32, D_MODEL / 32, 4), 256, 0, stream>>>(wp);

  QKVPtrs p;
  p.WT[0] = wqT; p.WT[1] = wkT; p.WT[2] = wvT;
  p.bias[0] = bq; p.bias[1] = bk; p.bias[2] = bv;
  p.out[0] = qh; p.out[1] = kh; p.out[2] = vt;
  gemm_qkv<<<dim3(BS / 128, D_MODEL / 64, 3), 256, 0, stream>>>(xh, p);

  flash_attn<<<dim3(SEQ / 128, BATCH * N_HEADS), 256, 0, stream>>>(qh, kh, vt, ohb);

  gemm_out<<<dim3(BS / 128, D_MODEL / 64), 256, 0, stream>>>(ohb, woT, bo, out);
}
